// MoEGate_15728170238344
// MI455X (gfx1250) — matrix operand order repaired and checked by reading, not compiled, not run on hardware
//
#include <hip/hip_runtime.h>
#include <math.h>

// ---------------------------------------------------------------------------
// MoE gate for MI455X (gfx1250).
//  Phase 1: logits = x @ W^T via v_wmma_f32_16x16x32_bf16 (f32 accumulate).
//    - x (268 MB HBM stream) staged into LDS with double-buffered
//      global_load_async_to_lds_b128 (ASYNCcnt / s_wait_asynccnt), loaded
//      once per workgroup instead of once per wave.
//    - W (4 MB) stays L2-resident, streamed with direct global_load_b128.
//  Phase 2: sigmoid+bias in LDS, grouped top-2 / top-4-group / top-8 select.
// ---------------------------------------------------------------------------

typedef __attribute__((ext_vector_type(16))) __bf16 v16bf;
typedef __attribute__((ext_vector_type(8)))  float  v8f;

#define HD   4096   // hidden dim
#define NE   256    // experts
#define NG   8      // groups
#define EPG  32     // experts per group
#define TPW  16     // tokens per workgroup (one M tile)
#define NT   4      // 16-expert N tiles per wave (4 waves x 4 = 256 experts)
#define KC   128    // K chunk staged in LDS per stage
#define KCP  132    // padded row stride (floats): bank = (4*row + j) % 64
#define BUFF (TPW * KCP)          // floats per stage buffer
#define NSTG (HD / KC)            // 32 stages

__global__ __launch_bounds__(128)
void moe_gate_kernel(const float* __restrict__ x,
                     const float* __restrict__ wgt,
                     const float* __restrict__ bias,
                     float* __restrict__ out_idx,
                     float* __restrict__ out_w) {
  __shared__ float xs[2 * BUFF];   // 16.5 KB double-buffered x tile
  __shared__ float sc[TPW * NE];   // 16 KB logits -> corrected scores
  __shared__ float sb[NE];         // bias

  const int tid  = threadIdx.x;
  const int wave = tid >> 5;
  const int lane = tid & 31;
  const int half = lane >> 4;      // 0: lanes 0-15, 1: lanes 16-31
  const int l16  = lane & 15;
  const int t0   = blockIdx.x * TPW;

  for (int i = tid; i < NE; i += 128) sb[i] = bias[i];

  // ---- async-copy geometry: stage = 16 rows x 128 floats = 512 x 16B chunks,
  //      4 chunks per thread; each wave issues exactly 4 async ops per stage.
  const int chunk_row = tid >> 5;          // reused per i: row = (tid+128i)>>5
  (void)chunk_row;

  // LDS byte offset of xs[] (flat shared pointer: offset lives in addr[31:0])
  const unsigned xs_base = (unsigned)(uintptr_t)(&xs[0]);

  auto issue_stage = [&](int s) {
    const int kb0      = s * KC;
    const unsigned bb  = xs_base + (unsigned)((s & 1) * BUFF * 4);
#pragma unroll
    for (int i = 0; i < 4; ++i) {
      const int chunk = tid + i * 128;     // 0..511
      const int row   = chunk >> 5;        // 32 x 16B chunks per row
      const int seg   = chunk & 31;
      const float*  gp   = x + (size_t)(t0 + row) * HD + kb0 + seg * 4;
      const unsigned loff = bb + (unsigned)(row * KCP + seg * 4) * 4u;
      asm volatile("global_load_async_to_lds_b128 %0, %1, off"
                   :: "v"(loff), "v"(gp) : "memory");
    }
  };

  // B: expert row; lane holds N = l16, K = kb + half*8 + 0..7 and +16..23.
  const float* wbase = wgt + (size_t)(wave * (NT * 16) + l16) * HD + half * 8;

  v8f acc[NT] = {};

  issue_stage(0);

  for (int s = 0; s < NSTG; ++s) {
    if (s + 1 < NSTG) {
      issue_stage(s + 1);
      // current stage (first 4 per-wave async ops) done; next stage in flight
      asm volatile("s_wait_asynccnt 0x4" ::: "memory");
    } else {
      asm volatile("s_wait_asynccnt 0x0" ::: "memory");
    }
    __syncthreads();   // make async-written tile visible to all waves

    const float* xbuf = &xs[(s & 1) * BUFF];
#pragma unroll
    for (int kbl = 0; kbl < KC; kbl += 32) {
      // A from LDS; per ISA 16-bit A layout: lanes 0-15 supply K = +0..7 and
      // +16..23, lanes 16-31 supply K = +8..15 and +24..31.
      const float* ar = xbuf + l16 * KCP + kbl + half * 8;
      float4 a0 = *(const float4*)(ar + 0);
      float4 a1 = *(const float4*)(ar + 4);
      float4 a2 = *(const float4*)(ar + 16);
      float4 a3 = *(const float4*)(ar + 20);

      v16bf A;
      A[0]  = (__bf16)a0.x; A[1]  = (__bf16)a0.y; A[2]  = (__bf16)a0.z; A[3]  = (__bf16)a0.w;
      A[4]  = (__bf16)a1.x; A[5]  = (__bf16)a1.y; A[6]  = (__bf16)a1.z; A[7]  = (__bf16)a1.w;
      A[8]  = (__bf16)a2.x; A[9]  = (__bf16)a2.y; A[10] = (__bf16)a2.z; A[11] = (__bf16)a2.w;
      A[12] = (__bf16)a3.x; A[13] = (__bf16)a3.y; A[14] = (__bf16)a3.z; A[15] = (__bf16)a3.w;

      const int kb = s * KC + kbl;
#pragma unroll
      for (int nt = 0; nt < NT; ++nt) {
        const float4* bp = (const float4*)(wbase + (size_t)nt * 16 * HD + kb);
        float4 b0 = bp[0], b1 = bp[1], b2 = bp[4], b3 = bp[5];
        v16bf B;
        B[0]  = (__bf16)b0.x; B[1]  = (__bf16)b0.y; B[2]  = (__bf16)b0.z; B[3]  = (__bf16)b0.w;
        B[4]  = (__bf16)b1.x; B[5]  = (__bf16)b1.y; B[6]  = (__bf16)b1.z; B[7]  = (__bf16)b1.w;
        B[8]  = (__bf16)b2.x; B[9]  = (__bf16)b2.y; B[10] = (__bf16)b2.z; B[11] = (__bf16)b2.w;
        B[12] = (__bf16)b3.x; B[13] = (__bf16)b3.y; B[14] = (__bf16)b3.z; B[15] = (__bf16)b3.w;

        acc[nt] = __builtin_amdgcn_wmma_f32_16x16x32_bf16(
            /*neg_a=*/false, A, /*neg_b=*/false, B,
            /*c_mod=*/(short)0, acc[nt], /*reuse_a=*/false, /*reuse_b=*/false);
      }
    }
    __syncthreads();   // all reads of this buffer done before it is re-filled
  }

  // C/D layout: lane L, vgpr r -> M = r + 8*(L/16), N = L%16.
#pragma unroll
  for (int nt = 0; nt < NT; ++nt) {
    const int e = wave * (NT * 16) + nt * 16 + l16;
#pragma unroll
    for (int r = 0; r < 8; ++r) {
      sc[(half * 8 + r) * NE + e] = acc[nt][r];
    }
  }
  __syncthreads();

  // corrected scores in place: sigmoid(logit) + bias
  for (int i = tid; i < TPW * NE; i += 128) {
    float lg = sc[i];
    float s  = 1.0f / (1.0f + __expf(-lg));
    sc[i] = s + sb[i & (NE - 1)];
  }
  __syncthreads();

  // one lane per token does the grouped top-k
  if (tid < TPW) {
    float* row = sc + tid * NE;
    const float NEGINF = -__builtin_inff();

    // group score = sum of top-2 corrected scores within each group of 32
    float gsc[NG];
#pragma unroll
    for (int g = 0; g < NG; ++g) {
      float m1 = NEGINF, m2 = NEGINF;
      for (int j = 0; j < EPG; ++j) {
        float v = row[g * EPG + j];
        if (v > m1)      { m2 = m1; m1 = v; }
        else if (v > m2) { m2 = v; }
      }
      gsc[g] = m1 + m2;
    }

    // top-4 groups (strict '>' => ties pick lowest index, like jax top_k)
    unsigned gmask = 0u;
    for (int k = 0; k < 4; ++k) {
      float best = NEGINF; int bg = 0;
      for (int g = 0; g < NG; ++g)
        if (!((gmask >> g) & 1u) && gsc[g] > best) { best = gsc[g]; bg = g; }
      gmask |= 1u << bg;
    }

    // top-8 experts in selected groups; ascending-index scan + -inf knockout
    // reproduces jax's descending-value / first-occurrence ordering.
    int   bi[8];
    float bw[8];
    float wsum = 0.0f;
    for (int k = 0; k < 8; ++k) {
      float best = NEGINF; int be = 0;
      for (int e = 0; e < NE; ++e) {
        if ((gmask >> (e >> 5)) & 1u) {
          float v = row[e];
          if (v > best) { best = v; be = e; }
        }
      }
      row[be] = NEGINF;
      float s = best - sb[be];   // recover raw sigmoid score
      bi[k] = be; bw[k] = s; wsum += s;
    }

    const float scale = 2.5f / (wsum + 1e-20f);
    const int t = t0 + tid;
#pragma unroll
    for (int k = 0; k < 8; ++k) {
      out_idx[t * 8 + k] = (float)bi[k];
      out_w  [t * 8 + k] = bw[k] * scale;
    }
  }
}

extern "C" void kernel_launch(void* const* d_in, const int* in_sizes, int n_in,
                              void* d_out, int out_size, void* d_ws, size_t ws_size,
                              hipStream_t stream) {
  const float* x    = (const float*)d_in[0];   // [4,4096,4096] f32
  const float* wgt  = (const float*)d_in[1];   // [256,4096] f32
  const float* bias = (const float*)d_in[2];   // [256] f32

  const int T = in_sizes[0] / HD;              // 16384 tokens
  float* out      = (float*)d_out;
  float* out_idx  = out;                       // first T*8: indices (as float)
  float* out_w    = out + (size_t)T * 8;       // next  T*8: weights

  dim3 grid(T / TPW);   // 1024
  dim3 block(128);      // 4 waves
  hipLaunchKernelGGL(moe_gate_kernel, grid, block, 0, stream,
                     x, wgt, bias, out_idx, out_w);
}
